// FullAttention_84911503442235
// MI455X (gfx1250) — matrix operand order repaired and checked by reading, not compiled, not run on hardware
//
#include <hip/hip_runtime.h>
#include <math.h>
#include <type_traits>

typedef __attribute__((ext_vector_type(16))) __bf16 v16bf;
typedef __attribute__((ext_vector_type(8)))  __bf16 v8bf;
typedef __attribute__((ext_vector_type(8)))  float  v8f;

#define B_ 4
#define L_ 2048
#define H_ 16
#define E_ 64
#define RS (H_*E_)            // float stride between consecutive sequence positions
#define SCALE 0.125f          // 1/sqrt(64)
#define NEGC (-1.0e6f)
#define POSC ( 1.0e6f)
#define WAVES 8
#define MTILE (WAVES*16)      // 128 query rows per workgroup
#define SBLK 64               // keys per softmax pass

#define KPAD 72               // bf16 per K-tile row (64 data + 8 pad) -> 144B stride
#define VPAD 72               // bf16 per Vt row    (64 data + 8 pad) -> 144B stride
#define PPAD 72               // bf16 per P row     (64 data + 8 pad) -> 144B stride

__global__ __launch_bounds__(256) void fa_causal_bf16_wmma(
    const float* __restrict__ Q,
    const float* __restrict__ Kp,
    const float* __restrict__ Vp,
    float* __restrict__ out)
{
  __shared__ __align__(16) __bf16 Kbuf[2][64][KPAD];   // [buf][key][e]
  __shared__ __align__(16) __bf16 Vt[2][64][VPAD];     // [buf][e][key] (transposed)
  __shared__ __align__(16) __bf16 Pbuf[WAVES][16][PPAD]; // per-wave P, columns permuted

  const int tid  = threadIdx.x;
  const int lane = tid & 31;
  const int w    = tid >> 5;
  const int half = lane >> 4;
  const int l16  = lane & 15;

  const int qt = blockIdx.x;
  const int h  = blockIdx.y;
  const int b  = blockIdx.z;

  const int q0 = qt * MTILE + w * 16;    // first query row of this wave

  const float* Qb = Q  + ((size_t)b * L_ * H_ + h) * E_;
  const float* Kb = Kp + ((size_t)b * L_ * H_ + h) * E_;
  const float* Vb = Vp + ((size_t)b * L_ * H_ + h) * E_;
  float*       Ob = out + ((size_t)b * L_ * H_ + h) * E_;

  // ---- Q A-fragments (16x32 bf16 each, two E-chunks of 32), loaded once ----
  v16bf qa[2];
  {
    const float* qrow = Qb + (size_t)(q0 + l16) * RS;
    #pragma unroll
    for (int c = 0; c < 2; ++c) {
      #pragma unroll
      for (int rb = 0; rb < 2; ++rb) {
        const int e = 32*c + 16*half + 8*rb;
        const float4* p = reinterpret_cast<const float4*>(qrow + e);
        float4 x0 = p[0], x1 = p[1];
        const int fb = 8*rb;
        qa[c][fb+0]=(__bf16)x0.x; qa[c][fb+1]=(__bf16)x0.y;
        qa[c][fb+2]=(__bf16)x0.z; qa[c][fb+3]=(__bf16)x0.w;
        qa[c][fb+4]=(__bf16)x1.x; qa[c][fb+5]=(__bf16)x1.y;
        qa[c][fb+6]=(__bf16)x1.z; qa[c][fb+7]=(__bf16)x1.w;
      }
    }
  }

  // all-ones B fragment: every output column of P*ones = row-sum of P
  v16bf ones;
  #pragma unroll
  for (int i = 0; i < 16; ++i) ones[i] = (__bf16)1.0f;

  // ---- cooperative staging of one 64-key block into LDS (bf16) ----
  auto stage = [&](int kbase, int bufi) {
    {
      // K: 64 rows x 64 e, row-major. 4096 elems / 256 threads = 16 each.
      const int row = tid >> 2;
      const int e0  = (tid & 3) * 16;
      const float4* p = reinterpret_cast<const float4*>(Kb + (size_t)(kbase + row) * RS + e0);
      float4 x0 = p[0], x1 = p[1], x2 = p[2], x3 = p[3];
      v8bf d0, d1;
      d0[0]=(__bf16)x0.x; d0[1]=(__bf16)x0.y; d0[2]=(__bf16)x0.z; d0[3]=(__bf16)x0.w;
      d0[4]=(__bf16)x1.x; d0[5]=(__bf16)x1.y; d0[6]=(__bf16)x1.z; d0[7]=(__bf16)x1.w;
      d1[0]=(__bf16)x2.x; d1[1]=(__bf16)x2.y; d1[2]=(__bf16)x2.z; d1[3]=(__bf16)x2.w;
      d1[4]=(__bf16)x3.x; d1[5]=(__bf16)x3.y; d1[6]=(__bf16)x3.z; d1[7]=(__bf16)x3.w;
      *reinterpret_cast<v8bf*>(&Kbuf[bufi][row][e0])     = d0;
      *reinterpret_cast<v8bf*>(&Kbuf[bufi][row][e0 + 8]) = d1;
    }
    {
      // V: transpose into Vt[e][s]. Thread owns (e, 16-key run).
      const int e  = tid & 63;
      const int s0 = (tid >> 6) * 16;
      const float* src = Vb + (size_t)(kbase + s0) * RS + e;
      v8bf d0, d1;
      #pragma unroll
      for (int i = 0; i < 8; ++i) {
        d0[i] = (__bf16) src[(size_t)i * RS];
        d1[i] = (__bf16) src[(size_t)(i + 8) * RS];
      }
      *reinterpret_cast<v8bf*>(&Vt[bufi][e][s0])     = d0;
      *reinterpret_cast<v8bf*>(&Vt[bufi][e][s0 + 8]) = d1;
    }
  };

  // B-fragment of K^T from LDS: lane -> key column, elem f -> e = 32*ec + 16*half + f
  auto kfrag = [&](int bufi, int sub, int ec) -> v16bf {
    const __bf16* p = &Kbuf[bufi][sub + l16][32*ec + 16*half];
    v8bf lo = *reinterpret_cast<const v8bf*>(p);
    v8bf hi = *reinterpret_cast<const v8bf*>(p + 8);
    v16bf f;
    #pragma unroll
    for (int i = 0; i < 8; ++i) { f[i] = lo[i]; f[8+i] = hi[i]; }
    return f;
  };

  // B-fragment of V from transposed LDS: lane -> out col e, elem f<8 -> key = 32*kc + 8*half + f, f>=8 -> key = 32*kc + 16 + 8*half + (f-8)
  auto vfrag = [&](int bufi, int nt, int kc) -> v16bf {
    const __bf16* p = &Vt[bufi][16*nt + l16][32*kc + 8*half];
    v8bf lo = *reinterpret_cast<const v8bf*>(p);
    v8bf hi = *reinterpret_cast<const v8bf*>(p + 16);
    v16bf f;
    #pragma unroll
    for (int i = 0; i < 8; ++i) { f[i] = lo[i]; f[8+i] = hi[i]; }
    return f;
  };

  // o[0..3]: output tiles, o[4]: row-sum accumulator (flash "l")
  v8f o[5];
  #pragma unroll
  for (int nt = 0; nt < 5; ++nt) o[nt] = (v8f)0.0f;
  float mrow[8];
  #pragma unroll
  for (int r = 0; r < 8; ++r) mrow[r] = -__builtin_inff();

  const int kbmaxwg = qt * 2 + 1;        // (qt*128+127)>>6
  const int kbmax   = (q0 + 15) >> 6;    // this wave's causal limit (inclusive)

  stage(0, 0);

  for (int kb = 0; kb <= kbmaxwg; ++kb) {
    __syncthreads();

    if (kb + 1 <= kbmaxwg)
      stage((kb + 1) << 6, (kb + 1) & 1);

    if (kb + 2 <= kbmaxwg) {
      const int pk = (kb + 2) << 6;
      __builtin_prefetch(Kb + (size_t)(pk + (tid >> 2)) * RS + (tid & 3) * 16, 0, 1);
      __builtin_prefetch(Vb + (size_t)(pk + (tid >> 6) * 16) * RS + (tid & 63), 0, 1);
    }

    if (kb <= kbmax) {                   // wave-uniform causal skip (EXEC stays full)
      const int bufi  = kb & 1;
      const int kbase = kb << 6;

      // ---- scores S = Q K^T : four 16x16 fp32 tiles (64 keys) ----
      v8f s[4];
      #pragma unroll
      for (int j = 0; j < 4; ++j) {
        v8f acc = (v8f)0.0f;
        v16bf f = kfrag(bufi, 16*j, 0);
        acc = __builtin_amdgcn_wmma_f32_16x16x32_bf16(false, qa[0], false, f, (short)0, acc, false, false);
        f = kfrag(bufi, 16*j, 1);
        acc = __builtin_amdgcn_wmma_f32_16x16x32_bf16(false, qa[1], false, f, (short)0, acc, false, false);
        s[j] = acc;
      }

      const int rowb = q0 + 8*half;

      // softmax over 64 columns; masking only instantiated for diagonal blocks
      auto softmax_rows = [&](auto domask_c) {
        constexpr bool domask = decltype(domask_c)::value;
        #pragma unroll
        for (int r = 0; r < 8; ++r) {
          const int row = rowb + r;
          float sv[4];
          #pragma unroll
          for (int j = 0; j < 4; ++j) {
            float x = fminf(fmaxf(s[j][r], NEGC), POSC);
            if (domask) {
              const int col = kbase + 16*j + l16;
              x = (col <= row) ? x : NEGC;   // mask -> -inf -> clamp -> -1e6
            }
            sv[j] = x * SCALE;
          }

          float bm = fmaxf(fmaxf(sv[0], sv[1]), fmaxf(sv[2], sv[3]));
          #pragma unroll
          for (int off = 1; off < 16; off <<= 1)
            bm = fmaxf(bm, __shfl_xor(bm, off, 32));   // within 16-lane half

          const float mnew  = fmaxf(mrow[r], bm);
          const float alpha = __expf(mrow[r] - mnew);
          mrow[r] = mnew;
          #pragma unroll
          for (int nt = 0; nt < 5; ++nt) o[nt][r] *= alpha;

          // columns permuted: c' = 4*(c%16) + c/16 -> lane's 4 values contiguous
          union { __bf16 hv[4]; unsigned long long u64; } pk;
          #pragma unroll
          for (int j = 0; j < 4; ++j) pk.hv[j] = (__bf16)__expf(sv[j] - mnew);
          const int rl = r + 8*half;
          *reinterpret_cast<unsigned long long*>(&Pbuf[w][rl][4*l16]) = pk.u64;
        }
      };
      if (kbase + SBLK - 1 > q0) softmax_rows(std::true_type{});
      else                       softmax_rows(std::false_type{});

      // order the cross-lane LDS store -> load within the wave
      __builtin_amdgcn_fence(__ATOMIC_RELEASE, "wavefront");
      __builtin_amdgcn_wave_barrier();
      __builtin_amdgcn_fence(__ATOMIC_ACQUIRE, "wavefront");

      // ---- re-read P as two A-fragments (16x32 bf16 each) ----
      // lane reads 64B span; permutation puts elems (j, j+8) of chunk cc in dword 2j+cc
      v16bf pa0, pa1;
      {
        const __bf16* pr = &Pbuf[w][l16][32*half];
        unsigned int dw[16];
        #pragma unroll
        for (int v = 0; v < 4; ++v) {
          uint4 x = reinterpret_cast<const uint4*>(pr)[v];
          dw[4*v+0] = x.x; dw[4*v+1] = x.y; dw[4*v+2] = x.z; dw[4*v+3] = x.w;
        }
        union U { unsigned int u[8]; v16bf v; } a0, a1;
        #pragma unroll
        for (int v = 0; v < 4; ++v) {
          const unsigned x0 = dw[4*v + 0], y0 = dw[4*v + 2];  // chunk 0
          const unsigned x1 = dw[4*v + 1], y1 = dw[4*v + 3];  // chunk 1
          a0.u[v]     = (x0 & 0xffffu) | (y0 << 16);
          a0.u[v + 4] = (x0 >> 16)     | (y0 & 0xffff0000u);
          a1.u[v]     = (x1 & 0xffffu) | (y1 << 16);
          a1.u[v + 4] = (x1 >> 16)     | (y1 & 0xffff0000u);
        }
        pa0 = a0.v; pa1 = a1.v;
      }
      __builtin_amdgcn_wave_barrier();

      // ---- O += P * V over K=64 ;  l += P * ones ----
      #pragma unroll
      for (int nt = 0; nt < 4; ++nt) {
        v16bf vb = vfrag(bufi, nt, 0);
        o[nt] = __builtin_amdgcn_wmma_f32_16x16x32_bf16(false, pa0, false, vb, (short)0, o[nt], false, false);
        vb = vfrag(bufi, nt, 1);
        o[nt] = __builtin_amdgcn_wmma_f32_16x16x32_bf16(false, pa1, false, vb, (short)0, o[nt], false, false);
      }
      o[4] = __builtin_amdgcn_wmma_f32_16x16x32_bf16(false, pa0, false, ones, (short)0, o[4], false, false);
      o[4] = __builtin_amdgcn_wmma_f32_16x16x32_bf16(false, pa1, false, ones, (short)0, o[4], false, false);
    }
  }

  // ---- normalize and store; o[4][r] holds the row sum in every lane ----
  #pragma unroll
  for (int r = 0; r < 8; ++r) {
    const int row = q0 + 8*half + r;
    const float inv = 1.0f / o[4][r];
    float* orow = Ob + (size_t)row * RS;
    #pragma unroll
    for (int nt = 0; nt < 4; ++nt)
      orow[16*nt + l16] = o[nt][r] * inv;
  }
}

extern "C" void kernel_launch(void* const* d_in, const int* in_sizes, int n_in,
                              void* d_out, int out_size, void* d_ws, size_t ws_size,
                              hipStream_t stream) {
  (void)in_sizes; (void)n_in; (void)out_size; (void)d_ws; (void)ws_size;
  const float* Q = (const float*)d_in[0];
  const float* K = (const float*)d_in[1];
  const float* V = (const float*)d_in[2];
  float* out = (float*)d_out;
  dim3 grid(L_ / MTILE, H_, B_);
  fa_causal_bf16_wmma<<<grid, dim3(256), 0, stream>>>(Q, K, V, out);
}
